// Memory_52252572123193
// MI455X (gfx1250) — matrix operand order repaired and checked by reading, not compiled, not run on hardware
//
#include <hip/hip_runtime.h>
#include <math.h>

// Problem constants (from reference setup_inputs)
#define BS      32
#define CH      512          // K dimension
#define HW      196          // 14*14
#define MROWS   (BS*HW)      // 6272 = 392 * 16
#define NKEYS   16384        // N dimension
#define NCLS    100
#define ALPHA   0.1f
#define MTILES  (MROWS/16)   // 392
#define SPLIT_SCALE 2048.0f

typedef _Float16 h4  __attribute__((ext_vector_type(4)));
typedef _Float16 h8  __attribute__((ext_vector_type(8)));
typedef _Float16 v16h __attribute__((ext_vector_type(16)));
typedef float    v8f  __attribute__((ext_vector_type(8)));

static __device__ inline v16h ld16(const _Float16* p0, const _Float16* p1) {
    h8 a = *(const h8*)p0;
    h8 b = *(const h8*)p1;
    return __builtin_shufflevector(a, b, 0,1,2,3,4,5,6,7,8,9,10,11,12,13,14,15);
}

static __device__ inline v8f wmma_f16(v16h a, v16h b, v8f c) {
    // (neg_a, A, neg_b, B, c_mod, C, reuse_a, reuse_b)
    return __builtin_amdgcn_wmma_f32_16x16x32_f16(false, a, false, b, (short)0, c, false, false);
}

// ---------------------------------------------------------------------------
// Kernel 1: L2-normalize queries over channel dim, emit f16 hi/lo split,
// transposed to row-major [M][512] (row = b*196 + loc).
// ---------------------------------------------------------------------------
__global__ void norm_split_q(const float* __restrict__ queries,
                             _Float16* __restrict__ qh, _Float16* __restrict__ ql) {
    __shared__ float red[256];
    const int row = blockIdx.x;
    const int b = row / HW, loc = row % HW;
    const int tid = threadIdx.x;
    const float* base = queries + (size_t)b * CH * HW + loc;
    float v0 = base[(size_t)tid * HW];
    float v1 = base[(size_t)(tid + 256) * HW];
    red[tid] = v0 * v0 + v1 * v1;
    __syncthreads();
    for (int s = 128; s > 0; s >>= 1) {
        if (tid < s) red[tid] += red[tid + s];
        __syncthreads();
    }
    const float inv = 1.0f / fmaxf(sqrtf(red[0]), 1e-12f);
    const float w0 = v0 * inv, w1 = v1 * inv;
    const size_t o = (size_t)row * CH;
    _Float16 h0 = (_Float16)w0;
    _Float16 h1 = (_Float16)w1;
    qh[o + tid]       = h0;
    ql[o + tid]       = (_Float16)((w0 - (float)h0) * SPLIT_SCALE);
    qh[o + tid + 256] = h1;
    ql[o + tid + 256] = (_Float16)((w1 - (float)h1) * SPLIT_SCALE);
}

// ---------------------------------------------------------------------------
// Kernel 2: split m_keys (already normalized) into f16 hi/lo, [N][512].
// ---------------------------------------------------------------------------
__global__ void split_keys(const float4* __restrict__ mk,
                           h4* __restrict__ kh, h4* __restrict__ kl) {
    const int i = blockIdx.x * 256 + threadIdx.x;   // exactly N*512/4 threads
    float4 v = mk[i];
    h4 hi, lo;
    hi[0] = (_Float16)v.x; lo[0] = (_Float16)((v.x - (float)hi[0]) * SPLIT_SCALE);
    hi[1] = (_Float16)v.y; lo[1] = (_Float16)((v.y - (float)hi[1]) * SPLIT_SCALE);
    hi[2] = (_Float16)v.z; lo[2] = (_Float16)((v.z - (float)hi[2]) * SPLIT_SCALE);
    hi[3] = (_Float16)v.w; lo[3] = (_Float16)((v.w - (float)hi[3]) * SPLIT_SCALE);
    kh[i] = hi;
    kl[i] = lo;
}

// ---------------------------------------------------------------------------
// Kernel 3: fused WMMA GEMM (6272x16384x512) + per-row argmax / masked maxes.
// One workgroup per 16-row M-tile; 8 waves split the N dimension.
// A tile staged in LDS (hi+lo), B streamed from L2.
// sims = Ch + Cl/2048  where Ch += Ah*Bh, Cl += Ah*Bl + Al*Bh.
// ---------------------------------------------------------------------------
__global__ void __launch_bounds__(256)
gemm_reduce(const _Float16* __restrict__ qh, const _Float16* __restrict__ ql,
            const _Float16* __restrict__ kh, const _Float16* __restrict__ kl,
            const int* __restrict__ m_vals, const int* __restrict__ targets,
            int* __restrict__ resVal, float* __restrict__ resPos,
            float* __restrict__ resNeg) {
    __shared__ __align__(16) _Float16 sA[2][16 * CH];   // 32 KB: [0]=hi [1]=lo
    __shared__ float sSim[8][16];
    __shared__ int   sIdx[8][16];
    __shared__ float sPos[8][16];
    __shared__ float sNeg[8][16];

    const int tid  = threadIdx.x;
    const int tile = blockIdx.x;

    // Cooperative contiguous copy of the 16-row A tile (hi and lo).
    {
        const uint4* s0 = (const uint4*)(qh + (size_t)tile * 16 * CH);
        const uint4* s1 = (const uint4*)(ql + (size_t)tile * 16 * CH);
        uint4* d0 = (uint4*)&sA[0][0];
        uint4* d1 = (uint4*)&sA[1][0];
        for (int i = tid; i < 1024; i += 256) { d0[i] = s0[i]; d1[i] = s1[i]; }
    }
    __syncthreads();

    const int wave  = tid >> 5;
    const int lane  = tid & 31;
    const int lrow  = lane & 15;            // A row / B col inside the tile
    const int half  = lane >> 4;
    const int abase = half ? 8 : 0;         // 16-bit A frag: K in {b..b+7} U {b+16..b+23}
    const int bbase = half ? 8 : 0;         // 16-bit B frag: K in {b..b+7} U {b+16..b+23}

    int tgt[8];
    #pragma unroll
    for (int r = 0; r < 8; ++r) {
        const int gm = tile * 16 + r + 8 * half;    // global M row for C-VGPR r
        tgt[r] = targets[gm / HW];
    }

    float bsim[8], pos_[8], neg_[8];
    int bidx[8];
    #pragma unroll
    for (int r = 0; r < 8; ++r) {
        bsim[r] = -3.4e38f; bidx[r] = 0x7fffffff; pos_[r] = 0.0f; neg_[r] = 0.0f;
    }

    for (int it = 0; it < NKEYS / 128; ++it) {      // 128 iterations
        const int n0   = it * 128 + wave * 16;
        const int ncol = n0 + lrow;                 // this lane's N column
        const _Float16* bh_p = kh + (size_t)ncol * CH;
        const _Float16* bl_p = kl + (size_t)ncol * CH;
        v8f ch = {};
        v8f cl = {};
        #pragma unroll 4
        for (int ks = 0; ks < CH / 32; ++ks) {      // 16 K-steps of 32
            const int k0 = ks * 32;
            const _Float16* ah_p = &sA[0][lrow * CH + k0 + abase];
            const _Float16* al_p = &sA[1][lrow * CH + k0 + abase];
            v16h Ah = ld16(ah_p, ah_p + 16);
            v16h Al = ld16(al_p, al_p + 16);
            v16h Bh = ld16(bh_p + k0 + bbase, bh_p + k0 + bbase + 16);
            v16h Bl = ld16(bl_p + k0 + bbase, bl_p + k0 + bbase + 16);
            ch = wmma_f16(Ah, Bh, ch);
            cl = wmma_f16(Ah, Bl, cl);
            cl = wmma_f16(Al, Bh, cl);
        }
        const int mv = m_vals[ncol];
        #pragma unroll
        for (int r = 0; r < 8; ++r) {
            const float s = ch[r] + cl[r] * (1.0f / SPLIT_SCALE);
            if (s > bsim[r]) { bsim[r] = s; bidx[r] = ncol; }   // strict > keeps lowest n
            if (mv == tgt[r]) pos_[r] = fmaxf(pos_[r], s);
            else              neg_[r] = fmaxf(neg_[r], s);
        }
    }

    // Cross-lane reduction: lanes 0-15 own row r, lanes 16-31 own row r+8.
    #pragma unroll
    for (int r = 0; r < 8; ++r) {
        float s = bsim[r]; int ix = bidx[r];
        float p = pos_[r]; float ng = neg_[r];
        #pragma unroll
        for (int m = 8; m >= 1; m >>= 1) {
            float os  = __shfl_xor(s, m, 32);
            int   oix = __shfl_xor(ix, m, 32);
            float op  = __shfl_xor(p, m, 32);
            float ong = __shfl_xor(ng, m, 32);
            if (os > s || (os == s && oix < ix)) { s = os; ix = oix; }
            p  = fmaxf(p, op);
            ng = fmaxf(ng, ong);
        }
        if ((lane & 15) == 0) {
            const int rr = r + 8 * half;
            sSim[wave][rr] = s; sIdx[wave][rr] = ix;
            sPos[wave][rr] = p; sNeg[wave][rr] = ng;
        }
    }
    __syncthreads();

    // Combine 8 wave partials, resolve nearest slot -> m_vals, write results.
    if (tid < 16) {
        float s = sSim[0][tid]; int ix = sIdx[0][tid];
        float p = sPos[0][tid]; float ng = sNeg[0][tid];
        #pragma unroll
        for (int w = 1; w < 8; ++w) {
            const float os = sSim[w][tid]; const int oix = sIdx[w][tid];
            if (os > s || (os == s && oix < ix)) { s = os; ix = oix; }
            p  = fmaxf(p, sPos[w][tid]);
            ng = fmaxf(ng, sNeg[w][tid]);
        }
        const int gm = tile * 16 + tid;
        resVal[gm] = m_vals[ix];
        resPos[gm] = p;
        resNeg[gm] = ng;
    }
}

// ---------------------------------------------------------------------------
// Kernel 4: per-batch majority vote (jnp.argmax tie-break: lowest class) and
// per-batch mean triplet loss. One block per batch sample.
// ---------------------------------------------------------------------------
__global__ void vote_loss(const int* __restrict__ resVal,
                          const float* __restrict__ resPos,
                          const float* __restrict__ resNeg,
                          const int* __restrict__ m_vals,
                          const int* __restrict__ targets,
                          float* __restrict__ d_out,
                          float* __restrict__ lossPart) {
    __shared__ int counts[NCLS];
    __shared__ int flag;
    __shared__ float lred[256];
    const int b = blockIdx.x, tid = threadIdx.x;
    if (tid < NCLS) counts[tid] = 0;
    if (tid == 0) flag = 0;
    __syncthreads();
    const int tg = targets[b];
    for (int i = tid; i < NKEYS; i += 256)
        if (m_vals[i] == tg) flag = 1;              // benign write race (only 1s)
    for (int l = tid; l < HW; l += 256)
        atomicAdd(&counts[resVal[b * HW + l]], 1);
    __syncthreads();
    const int hp = flag;
    float acc = 0.0f;
    for (int l = tid; l < HW; l += 256) {
        const float p = resPos[b * HW + l], ng = resNeg[b * HW + l];
        const float t = hp ? (ng - p + ALPHA) : (ng + ALPHA);
        acc += fmaxf(t, 0.0f);
    }
    lred[tid] = acc;
    __syncthreads();
    for (int s = 128; s > 0; s >>= 1) {
        if (tid < s) lred[tid] += lred[tid + s];
        __syncthreads();
    }
    if (tid == 0) {
        lossPart[b] = lred[0] / (float)HW;
        int best = 0, bc = counts[0];
        for (int c = 1; c < NCLS; ++c)
            if (counts[c] > bc) { bc = counts[c]; best = c; }   // strict >: first max
        d_out[b] = (float)best;
    }
}

// Kernel 5: sum of per-batch losses -> d_out[32]
__global__ void finalize(const float* __restrict__ lossPart, float* __restrict__ d_out) {
    if (threadIdx.x == 0) {
        float s = 0.0f;
        for (int i = 0; i < BS; ++i) s += lossPart[i];
        d_out[BS] = s;
    }
}

// ---------------------------------------------------------------------------
extern "C" void kernel_launch(void* const* d_in, const int* in_sizes, int n_in,
                              void* d_out, int out_size, void* d_ws, size_t ws_size,
                              hipStream_t stream) {
    (void)in_sizes; (void)n_in; (void)out_size; (void)ws_size;
    const float* queries = (const float*)d_in[0];
    const int*   targets = (const int*)d_in[1];
    const float* m_keys  = (const float*)d_in[2];
    const int*   m_vals  = (const int*)d_in[3];
    float* out = (float*)d_out;
    char* ws = (char*)d_ws;

    // Workspace layout (~46.5 MB total)
    const size_t qbytes = (size_t)MROWS * CH * sizeof(_Float16);   // 6.42 MB
    const size_t kbytes = (size_t)NKEYS * CH * sizeof(_Float16);   // 16.8 MB
    _Float16* qh = (_Float16*)(ws);
    _Float16* ql = (_Float16*)(ws + qbytes);
    _Float16* kh = (_Float16*)(ws + 2 * qbytes);
    _Float16* kl = (_Float16*)(ws + 2 * qbytes + kbytes);
    char* tail = ws + 2 * qbytes + 2 * kbytes;
    int*   resVal   = (int*)tail;
    float* resPos   = (float*)(tail + (size_t)MROWS * 4);
    float* resNeg   = (float*)(tail + (size_t)MROWS * 8);
    float* lossPart = (float*)(tail + (size_t)MROWS * 12);

    norm_split_q<<<MROWS, 256, 0, stream>>>(queries, qh, ql);
    split_keys<<<(NKEYS * CH / 4) / 256, 256, 0, stream>>>((const float4*)m_keys,
                                                           (h4*)kh, (h4*)kl);
    gemm_reduce<<<MTILES, 256, 0, stream>>>(qh, ql, kh, kl, m_vals, targets,
                                            resVal, resPos, resNeg);
    vote_loss<<<BS, 256, 0, stream>>>(resVal, resPos, resNeg, m_vals, targets,
                                      out, lossPart);
    finalize<<<1, 32, 0, stream>>>(lossPart, out);
}
